// CorrelationLayer_29008209117830
// MI455X (gfx1250) — matrix operand order repaired and checked by reading, not compiled, not run on hardware
//
#include <hip/hip_runtime.h>

// CDNA5 (gfx1250) correlation layer using v_wmma_f32_16x16x32_f16.
// out[b, (dy+4)*9+(dx+4), h, w] = sum_c f1[b,c,h,w] * f2[b,c,(h-dy)%H,(w-dx)%W]
//
// v4: __builtin_amdgcn_sched_barrier(0) pins the software pipeline: the
//     ds_load_b128 group for dy+1 must issue before the WMMAs of dy, so the
//     two B-fragment buffers get distinct registers and each WMMA pair waits
//     on dscnt<=4 (loads in flight) instead of a serial dscnt==0 round-trip.

typedef __attribute__((ext_vector_type(16))) _Float16 v16h;
typedef __attribute__((ext_vector_type(8)))  float    v8f;

namespace {
constexpr int Bn = 8, Cn = 512, Hn = 64, Wn = 128;
constexpr int ND = 9;           // displacements per axis
constexpr int NDISP = 81;
constexpr int KC = 32;          // K per WMMA chunk (16 packed f16x2 dwords)
constexpr int WT = 16;          // pixel tile width (M dim)
constexpr int HT = 4;           // h rows per block (1 per wave)
constexpr int ROWS = HT + 8;    // f2 halo rows: hb-4 .. hb+7
constexpr int COLS = 32;        // f2 halo cols: w0-4 .. w0+27 (two N tiles)
constexpr int CS = 20;          // dwords per column (16 data + 4 pad):
                                //   16B-aligned, n*20 mod 64 tiles all banks

union H2U { _Float16 h[2]; unsigned int u; };
union Frag { v16h v; uint4 q[2]; };

// Unused: A and B are stored with identity k2 slots; posA's slot order per lane-half is
// hi=0 -> k2 {0,1,2,3, 8,9,10,11}, hi=1 -> k2 {4,5,6,7, 12,13,14,15}.
__device__ __forceinline__ int posA(int k2) {
  return (k2 & 3) | (((k2 >> 2) & 1) << 3) | ((k2 >> 3) << 2);
}
}

__global__ __launch_bounds__(128, 1)
void corr_wmma_kernel(const float* __restrict__ f1,
                      const float* __restrict__ f2,
                      float* __restrict__ out) {
  // [row][col][k2-slot], packed f16x2 dwords, 16B aligned
  __shared__ uint4 ldsBq[ROWS * COLS * CS / 4];   // 30 KB
  __shared__ uint4 ldsAq[HT * WT * CS / 4];       //  5 KB
  unsigned int* ldsB = (unsigned int*)ldsBq;
  unsigned int* ldsA = (unsigned int*)ldsAq;

  const int tid  = threadIdx.x;
  const int wv   = tid >> 5;        // wave id = local h row
  const int lane = tid & 31;
  const int n    = lane & 15;       // N-column / M-row within half
  const int hi   = lane >> 4;       // lane half selects K interleave / M offset

  const int w0 = blockIdx.x * WT;
  const int hb = blockIdx.y * HT;
  const int b  = blockIdx.z;

  const size_t planeHW = (size_t)Hn * Wn;

  // 18 fp32 accumulators: 9 dy rows x 2 N-tiles (cols [0,16) and [16,32))
  v8f acc[ND][2];
  #pragma unroll
  for (int d = 0; d < ND; ++d)
    #pragma unroll
    for (int t = 0; t < 2; ++t)
      #pragma unroll
      for (int i = 0; i < 8; ++i) acc[d][t][i] = 0.0f;

  // Per-lane LDS fragment addresses (loop-invariant)
  const unsigned int* aPtr = ldsA + wv * (WT * CS) + n * CS + hi * 8;
  const int bOff0 = n * CS + hi * 8;
  const int bOff1 = (n + 16) * CS + hi * 8;

  #pragma unroll 1
  for (int kc = 0; kc < Cn; kc += KC) {
    const bool pf = (kc + KC) < Cn;

    // ---- stage f2 halo: 12 rows x 16 k-pairs x 8 col-groups of 4 ----
    const float* f2b = f2 + ((size_t)b * Cn + kc) * planeHW;
    for (int s = tid; s < ROWS * 16 * 8; s += 128) {
      const int cg  = s & 7;
      const int k2  = (s >> 3) & 15;
      const int row = s >> 7;
      const int hh  = (hb - 4 + row) & (Hn - 1);       // circular roll in H
      const int wwb = (w0 - 4 + cg * 4) & (Wn - 1);    // circular roll in W (4-aligned)
      const float* p = f2b + (size_t)(2 * k2) * planeHW + (size_t)hh * Wn + wwb;
      const float4 lo = *(const float4*)p;
      const float4 hx = *(const float4*)(p + planeHW);
      if (pf && !(cg & 1)) {                           // one prefetch per cacheline
        __builtin_prefetch(p + (size_t)KC * planeHW, 0, 1);
        __builtin_prefetch(p + (size_t)(KC + 1) * planeHW, 0, 1);
      }
      const int base = row * (COLS * CS) + (cg * 4) * CS + k2;  // B: identity k2 slot
      const float* lp = (const float*)&lo;
      const float* hp = (const float*)&hx;
      #pragma unroll
      for (int i = 0; i < 4; ++i) {
        H2U pk;
        pk.h[0] = (_Float16)lp[i];
        pk.h[1] = (_Float16)hp[i];
        ldsB[base + i * CS] = pk.u;
      }
    }
    // ---- stage f1 tiles: 4 waves x 16 k-pairs x 4 pixel-groups of 4 ----
    const float* f1b = f1 + ((size_t)b * Cn + kc) * planeHW;
    for (int s = tid; s < HT * 16 * 4; s += 128) {
      const int g   = s & 3;
      const int k2  = (s >> 2) & 15;
      const int wr  = s >> 6;
      const float* p = f1b + (size_t)(2 * k2) * planeHW
                           + (size_t)(hb + wr) * Wn + (w0 + g * 4);
      const float4 lo = *(const float4*)p;
      const float4 hx = *(const float4*)(p + planeHW);
      if (pf && !(g & 1)) {
        __builtin_prefetch(p + (size_t)KC * planeHW, 0, 1);
        __builtin_prefetch(p + (size_t)(KC + 1) * planeHW, 0, 1);
      }
      const int base = wr * (WT * CS) + (g * 4) * CS + k2;
      const float* lp = (const float*)&lo;
      const float* hp = (const float*)&hx;
      #pragma unroll
      for (int i = 0; i < 4; ++i) {
        H2U pk;
        pk.h[0] = (_Float16)lp[i];
        pk.h[1] = (_Float16)hp[i];
        ldsA[base + i * CS] = pk.u;
      }
    }
    __syncthreads();

    // ---- A fragment (16x32 f16): two aligned ds_load_b128 ----
    Frag a;
    a.q[0] = ((const uint4*)aPtr)[0];
    a.q[1] = ((const uint4*)aPtr)[1];

    // ---- dy loop, B fragments double-buffered; sched_barrier(0) keeps the
    //      dy+1 loads ABOVE the dy WMMAs so the buffers cannot be coalesced
    //      and LDS latency overlaps matrix execution ----
    Frag bf[2][2];
    {
      const int rowb = (wv + 8) * (COLS * CS);          // dy = 0
      bf[0][0].q[0] = ((const uint4*)(ldsB + rowb + bOff0))[0];
      bf[0][0].q[1] = ((const uint4*)(ldsB + rowb + bOff0))[1];
      bf[0][1].q[0] = ((const uint4*)(ldsB + rowb + bOff1))[0];
      bf[0][1].q[1] = ((const uint4*)(ldsB + rowb + bOff1))[1];
    }
    #pragma unroll
    for (int dy = 0; dy < ND; ++dy) {
      const int cur = dy & 1;
      if (dy + 1 < ND) {
        const int nxt  = (dy + 1) & 1;
        const int rowb = (wv + 8 - (dy + 1)) * (COLS * CS);
        bf[nxt][0].q[0] = ((const uint4*)(ldsB + rowb + bOff0))[0];
        bf[nxt][0].q[1] = ((const uint4*)(ldsB + rowb + bOff0))[1];
        bf[nxt][1].q[0] = ((const uint4*)(ldsB + rowb + bOff1))[0];
        bf[nxt][1].q[1] = ((const uint4*)(ldsB + rowb + bOff1))[1];
      }
      __builtin_amdgcn_sched_barrier(0);   // loads(dy+1) stay above wmma(dy)
      acc[dy][0] = __builtin_amdgcn_wmma_f32_16x16x32_f16(
          false, a.v, false, bf[cur][0].v, (short)0, acc[dy][0], false, false);
      acc[dy][1] = __builtin_amdgcn_wmma_f32_16x16x32_f16(
          false, a.v, false, bf[cur][1].v, (short)0, acc[dy][1], false, false);
    }
    __syncthreads();
  }

  // ---- diagonal extraction + scatter store ----
  // G[m, j] with j = m + 8 - dxIdx; lane owns cols {n, n+16}, rows m = r + 8*hi.
  const int h = hb + wv;
  float* outb = out + (size_t)b * NDISP * planeHW;
  #pragma unroll
  for (int dy = 0; dy < ND; ++dy) {
    #pragma unroll
    for (int r = 0; r < 8; ++r) {
      const int m  = r + 8 * hi;
      const int j0 = m + ((n - m) & 15);       // unique j == n (mod 16) in [m, m+15]
      if (j0 <= m + 8) {                       // inside the 9-wide displacement band
        const int dx = m + 8 - j0;             // dxIdx in [0, 8]
        const float val = (j0 < 16) ? acc[dy][0][r] : acc[dy][1][r];
        const int d = dy * 9 + dx;
        outb[((size_t)d * Hn + h) * Wn + (w0 + m)] = val;
      }
    }
  }
}

extern "C" void kernel_launch(void* const* d_in, const int* in_sizes, int n_in,
                              void* d_out, int out_size, void* d_ws, size_t ws_size,
                              hipStream_t stream) {
  const float* f1 = (const float*)d_in[0];
  const float* f2 = (const float*)d_in[1];
  float* out = (float*)d_out;
  (void)in_sizes; (void)n_in; (void)out_size; (void)d_ws; (void)ws_size;

  dim3 grid(Wn / WT, Hn / HT, Bn);   // (8, 16, 8)
  dim3 block(128);                   // 4 waves: one h row each
  corr_wmma_kernel<<<grid, block, 0, stream>>>(f1, f2, out);
}
